// _PGCausalSelfAttention_58265526338197
// MI455X (gfx1250) — 8-bit matrix operand order repaired and checked by reading, compiled, not run on hardware
//
#include <hip/hip_runtime.h>
#include <hip/hip_bf16.h>
#include <math.h>
#include <stdint.h>

typedef int   v8i __attribute__((ext_vector_type(8)));
typedef float v8f __attribute__((ext_vector_type(8)));
typedef float v2f __attribute__((ext_vector_type(2)));

#define DIM    2048
#define SEQ    2048
#define BATCH  2
#define NHEADS 16
#define NKV    4
#define HD     128
#define KVD    512
#define FUSED  3072
#define NTOK   (BATCH*SEQ)

#define LDS_STRIDE 80

__device__ __forceinline__ void async_b128(unsigned lds_off, const void* gptr) {
  asm volatile("global_load_async_to_lds_b128 %0, %1, off"
               :: "v"(lds_off), "v"((uint64_t)(uintptr_t)gptr)
               : "memory");
}
__device__ __forceinline__ void wait_async3() {
  asm volatile("s_wait_asynccnt 0x3" ::: "memory");
}
__device__ __forceinline__ void wait_async0() {
  asm volatile("s_wait_asynccnt 0x0" ::: "memory");
}

__global__ void quant_rows_kernel(const float* __restrict__ src,
                                  signed char* __restrict__ dst,
                                  float* __restrict__ scale, int cols) {
  int row = blockIdx.x;
  const float* r = src + (size_t)row * cols;
  __shared__ float red[256];
  float mx = 0.f;
  for (int c = threadIdx.x; c < cols; c += 256) mx = fmaxf(mx, fabsf(r[c]));
  red[threadIdx.x] = mx;
  __syncthreads();
  for (int s = 128; s > 0; s >>= 1) {
    if (threadIdx.x < s) red[threadIdx.x] = fmaxf(red[threadIdx.x], red[threadIdx.x + s]);
    __syncthreads();
  }
  float sc = fmaxf(red[0] * (1.f / 127.f), 1e-5f);
  if (threadIdx.x == 0) scale[row] = sc;
  float inv = 1.f / sc;
  signed char* d = dst + (size_t)row * cols;
  for (int c = threadIdx.x; c < cols; c += 256) {
    float q = rintf(r[c] * inv);
    q = fminf(fmaxf(q, -127.f), 127.f);
    d[c] = (signed char)(int)q;
  }
}

__global__ void __launch_bounds__(256) gemm_i8_kernel(
    const signed char* __restrict__ Aq, const float* __restrict__ as_,
    const signed char* __restrict__ Wq, const float* __restrict__ ws_,
    float* __restrict__ out, int M, int N, int K) {
  __shared__ signed char Al[2][128 * LDS_STRIDE];
  __shared__ signed char Bl[2][64 * LDS_STRIDE];

  int tid  = threadIdx.x;
  int wave = tid >> 5, lane = tid & 31;
  int numNT = N >> 6;
  int bm = blockIdx.x / numNT;
  int bn = blockIdx.x - bm * numNT;
  int m = lane & 15, half = lane >> 4;

  const signed char* Abase = Aq + (size_t)(bm * 128) * K;
  const signed char* Bbase = Wq + (size_t)(bn * 64) * K;

  int ra0 = tid >> 2, seg = (tid & 3) << 4;
  int ra1 = (tid + 256) >> 2;
  int rb  = tid >> 2;

#define ISSUE_STAGE(buf, k0)                                                   \
  do {                                                                         \
    async_b128((unsigned)(uintptr_t)&Al[buf][ra0 * LDS_STRIDE + seg],          \
               Abase + (size_t)ra0 * K + (k0) + seg);                          \
    async_b128((unsigned)(uintptr_t)&Al[buf][ra1 * LDS_STRIDE + seg],          \
               Abase + (size_t)ra1 * K + (k0) + seg);                          \
    async_b128((unsigned)(uintptr_t)&Bl[buf][rb * LDS_STRIDE + seg],           \
               Bbase + (size_t)rb * K + (k0) + seg);                           \
  } while (0)

  ISSUE_STAGE(0, 0);

  v8i acc[4] = {};
  int nsteps = K >> 6;
  for (int s = 0; s < nsteps; ++s) {
    int buf = s & 1;
    if (s + 1 < nsteps) {
      ISSUE_STAGE(buf ^ 1, (s + 1) << 6);
      wait_async3();
    } else {
      wait_async0();
    }
    __syncthreads();

    const signed char* arow_l = &Al[buf][(wave * 16 + m) * LDS_STRIDE];
    v8i a;
#pragma unroll
    for (int v = 0; v < 8; ++v) {
      int kk = ((v >> 1) << 4) + (half << 3) + ((v & 1) << 2);
      a[v] = *(const int*)(arow_l + kk);
    }
    v8i b0, b1, b2, b3;
#pragma unroll
    for (int v = 0; v < 8; ++v) {
      int kb = ((v >> 1) << 4) + (half << 3) + ((v & 1) << 2);
      b0[v] = *(const int*)(&Bl[buf][(0 * 16 + m) * LDS_STRIDE] + kb);
      b1[v] = *(const int*)(&Bl[buf][(1 * 16 + m) * LDS_STRIDE] + kb);
      b2[v] = *(const int*)(&Bl[buf][(2 * 16 + m) * LDS_STRIDE] + kb);
      b3[v] = *(const int*)(&Bl[buf][(3 * 16 + m) * LDS_STRIDE] + kb);
    }
    acc[0] = __builtin_amdgcn_wmma_i32_16x16x64_iu8(true, a, true, b0, acc[0],
                                                    false, false);
    acc[1] = __builtin_amdgcn_wmma_i32_16x16x64_iu8(true, a, true, b1, acc[1],
                                                    false, false);
    acc[2] = __builtin_amdgcn_wmma_i32_16x16x64_iu8(true, a, true, b2, acc[2],
                                                    false, false);
    acc[3] = __builtin_amdgcn_wmma_i32_16x16x64_iu8(true, a, true, b3, acc[3],
                                                    false, false);

    __syncthreads();
  }
#undef ISSUE_STAGE

#pragma unroll
  for (int ss = 0; ss < 4; ++ss) {
    int col = (bn << 6) + (ss << 4) + m;
    float wsc = ws_[col];
#pragma unroll
    for (int v = 0; v < 8; ++v) {
      int row = bm * 128 + wave * 16 + v + (half << 3);
      out[(size_t)row * N + col] = (float)acc[ss][v] * as_[row] * wsc;
    }
  }
}

__global__ void rmsrope_kernel(float* __restrict__ qkv,
                               const float* __restrict__ q_gain) {
  int gid = (int)((blockIdx.x * blockDim.x + threadIdx.x) >> 5);
  int lane = threadIdx.x & 31;
  int t = gid / 20;
  int hv = gid - t * 20;
  if (t >= NTOK) return;

  float* base;
  float gain;
  if (hv < NHEADS) {
    base = qkv + (size_t)t * FUSED + hv * HD;
    gain = q_gain[hv];
  } else {
    base = qkv + (size_t)t * FUSED + DIM + (hv - NHEADS) * HD;
    gain = 1.f;
  }
  int p0 = lane, p1 = lane + 32;
  float x1a = base[p0], x2a = base[p0 + 64];
  float x1b = base[p1], x2b = base[p1 + 64];
  float ss = x1a * x1a + x2a * x2a + x1b * x1b + x2b * x2b;
#pragma unroll
  for (int msk = 16; msk >= 1; msk >>= 1) ss += __shfl_xor(ss, msk, 32);
  float r = rsqrtf(ss * (1.f / 128.f) + 1.1920929e-7f);

  int pos = t & (SEQ - 1);
  const float LN = 9.210340371976184f;
  float f0 = (float)pos * __expf(-((float)(2 * p0) * (1.f / 128.f)) * LN);
  float f1 = (float)pos * __expf(-((float)(2 * p1) * (1.f / 128.f)) * LN);
  float c0, s0, c1, s1;
  __sincosf(f0, &s0, &c0);
  __sincosf(f1, &s1, &c1);
  x1a *= r; x2a *= r; x1b *= r; x2b *= r;
  base[p0]      = (x1a * c0 + x2a * s0) * gain;
  base[p0 + 64] = (x2a * c0 - x1a * s0) * gain;
  base[p1]      = (x1b * c1 + x2b * s1) * gain;
  base[p1 + 64] = (x2b * c1 - x1b * s1) * gain;
}

__global__ void attn_kernel(const float* __restrict__ qkv, float* __restrict__ y) {
  int bid = blockIdx.x;
  int qt = bid & 127;
  int h  = (bid >> 7) & 15;
  int b  = bid >> 11;
  int lane = threadIdx.x;
  int m = lane & 15, half = lane >> 4;
  int kh = h >> 2;

  const float* qrow = qkv + (size_t)(b * SEQ + qt * 16 + m) * FUSED + h * HD;
  float qa[64];
#pragma unroll
  for (int kk = 0; kk < 32; ++kk) {
    qa[2 * kk]     = qrow[kk * 4 + half * 2];
    qa[2 * kk + 1] = qrow[kk * 4 + half * 2 + 1];
  }

  v8f o[8];
#pragma unroll
  for (int c = 0; c < 8; ++c) o[c] = (v8f){};
  float rmax[8], rsum[8];
#pragma unroll
  for (int v = 0; v < 8; ++v) { rmax[v] = -__builtin_inff(); rsum[v] = 0.f; }

  __shared__ float Pl[16][17];
  const float scale = 0.08838834764831845f;

  for (int j = 0; j <= qt; ++j) {
    const float* krow =
        qkv + (size_t)(b * SEQ + j * 16 + m) * FUSED + DIM + kh * HD;
    v8f sc = {};
#pragma unroll
    for (int kk = 0; kk < 32; ++kk) {
      v2f af; af.x = qa[2 * kk]; af.y = qa[2 * kk + 1];
      v2f bf; bf.x = krow[kk * 4 + half * 2]; bf.y = krow[kk * 4 + half * 2 + 1];
      sc = __builtin_amdgcn_wmma_f32_16x16x4_f32(false, af, false, bf, (short)0,
                                                 sc, false, false);
    }
    float corr[8];
#pragma unroll
    for (int v = 0; v < 8; ++v) {
      float e = sc[v] * scale;
      int mg = qt * 16 + v + half * 8;
      int ng = j * 16 + m;
      if (ng > mg) e = -3.0e38f;
      float rm = e;
      rm = fmaxf(rm, __shfl_xor(rm, 1, 32));
      rm = fmaxf(rm, __shfl_xor(rm, 2, 32));
      rm = fmaxf(rm, __shfl_xor(rm, 4, 32));
      rm = fmaxf(rm, __shfl_xor(rm, 8, 32));
      float nm = fmaxf(rmax[v], rm);
      float cf = __expf(rmax[v] - nm);
      float p = __expf(e - nm);
      float rs = p;
      rs += __shfl_xor(rs, 1, 32);
      rs += __shfl_xor(rs, 2, 32);
      rs += __shfl_xor(rs, 4, 32);
      rs += __shfl_xor(rs, 8, 32);
      rsum[v] = rsum[v] * cf + rs;
      rmax[v] = nm;
      corr[v] = cf;
      sc[v] = p;
    }
#pragma unroll
    for (int c = 0; c < 8; ++c)
#pragma unroll
      for (int v = 0; v < 8; ++v) o[c][v] *= corr[v];

    __syncthreads();
#pragma unroll
    for (int v = 0; v < 8; ++v) Pl[v + half * 8][m] = sc[v];
    __syncthreads();

    const float* vbase =
        qkv + (size_t)(b * SEQ + j * 16) * FUSED + DIM + KVD + kh * HD;
#pragma unroll
    for (int kc = 0; kc < 4; ++kc) {
      v2f pf;
      pf.x = Pl[m][kc * 4 + half * 2];
      pf.y = Pl[m][kc * 4 + half * 2 + 1];
      int kr0 = kc * 4 + half * 2;
      const float* vr0 = vbase + (size_t)kr0 * FUSED;
      const float* vr1 = vbase + (size_t)(kr0 + 1) * FUSED;
#pragma unroll
      for (int c = 0; c < 8; ++c) {
        v2f vf; vf.x = vr0[c * 16 + m]; vf.y = vr1[c * 16 + m];
        o[c] = __builtin_amdgcn_wmma_f32_16x16x4_f32(false, pf, false, vf,
                                                     (short)0, o[c], false, false);
      }
    }
  }

#pragma unroll
  for (int c = 0; c < 8; ++c) {
#pragma unroll
    for (int v = 0; v < 8; ++v) {
      int mr = v + half * 8;
      int tok = b * SEQ + qt * 16 + mr;
      y[(size_t)tok * DIM + h * HD + c * 16 + m] = o[c][v] / rsum[v];
    }
  }
}

extern "C" void kernel_launch(void* const* d_in, const int* in_sizes, int n_in,
                              void* d_out, int out_size, void* d_ws, size_t ws_size,
                              hipStream_t stream) {
  const float* x      = (const float*)d_in[0];
  const float* w_q    = (const float*)d_in[1];
  const float* w_k    = (const float*)d_in[2];
  const float* w_v    = (const float*)d_in[3];
  const float* w_o    = (const float*)d_in[4];
  const float* q_gain = (const float*)d_in[5];

  char* ws = (char*)d_ws;
  size_t off = 0;
  auto take = [&](size_t bytes) -> char* {
    char* p = ws + off;
    off = (off + bytes + 255) & ~(size_t)255;
    return p;
  };
  signed char* xq     = (signed char*)take((size_t)NTOK * DIM);
  float*       ascale = (float*)take((size_t)NTOK * 4);
  signed char* wq     = (signed char*)take((size_t)FUSED * DIM);
  float*       wscale = (float*)take((size_t)FUSED * 4);
  float*       qkv    = (float*)take((size_t)NTOK * FUSED * 4);
  float*       yb     = (float*)take((size_t)NTOK * DIM * 4);
  signed char* yq     = (signed char*)take((size_t)NTOK * DIM);
  float*       yscale = (float*)take((size_t)NTOK * 4);
  signed char* woq    = (signed char*)take((size_t)DIM * DIM);
  float*       wosc   = (float*)take((size_t)DIM * 4);
  (void)ws_size; (void)in_sizes; (void)n_in; (void)out_size;

  quant_rows_kernel<<<NTOK, 256, 0, stream>>>(x, xq, ascale, DIM);
  quant_rows_kernel<<<DIM, 256, 0, stream>>>(w_q, wq, wscale, DIM);
  quant_rows_kernel<<<KVD, 256, 0, stream>>>(w_k, wq + (size_t)DIM * DIM,
                                             wscale + DIM, DIM);
  quant_rows_kernel<<<KVD, 256, 0, stream>>>(w_v, wq + (size_t)(DIM + KVD) * DIM,
                                             wscale + DIM + KVD, DIM);
  gemm_i8_kernel<<<(NTOK / 128) * (FUSED / 64), 256, 0, stream>>>(
      xq, ascale, wq, wscale, qkv, NTOK, FUSED, DIM);
  rmsrope_kernel<<<NTOK * 20 / 8, 256, 0, stream>>>(qkv, q_gain);
  attn_kernel<<<BATCH * NHEADS * (SEQ / 16), 32, 0, stream>>>(qkv, yb);
  quant_rows_kernel<<<NTOK, 256, 0, stream>>>(yb, yq, yscale, DIM);
  quant_rows_kernel<<<DIM, 256, 0, stream>>>(w_o, woq, wosc, DIM);
  gemm_i8_kernel<<<(NTOK / 128) * (DIM / 64), 256, 0, stream>>>(
      yq, yscale, woq, wosc, (float*)d_out, NTOK, DIM, DIM);
}
